// DynamicWeightsCat11_28097676050578
// MI455X (gfx1250) — matrix operand order repaired and checked by reading, not compiled, not run on hardware
//
#include <hip/hip_runtime.h>
#include <hip/hip_bf16.h>

// ---------------------------------------------------------------------------
// Types for CDNA5 WMMA (wave32, v_wmma_f32_16x16x32_bf16)
// ---------------------------------------------------------------------------
typedef __bf16 bf16;
typedef __bf16 v16bf __attribute__((ext_vector_type(16)));
typedef __bf16 v8bf  __attribute__((ext_vector_type(8)));
typedef __bf16 v4bf  __attribute__((ext_vector_type(4)));
typedef float  v8f   __attribute__((ext_vector_type(8)));
typedef float  v4f   __attribute__((ext_vector_type(4)));

#define CH 512
#define CI 128
#define HW 128
#define NPIX 16384          // 128*128
#define NB 4                // batch
#define GROUPS 32

__device__ __constant__ int c_dils[4] = {1, 4, 8, 12};
__device__ __constant__ int c_ky[9] = {-1,-1,-1, 0,0,0, 1,1,1};
__device__ __constant__ int c_kx[9] = {-1, 0, 1,-1,0,1,-1,0,1};

__device__ inline bf16 f2bf(float f) {
  union { float f; unsigned u; } v; v.f = f;
  unsigned r = v.u + 0x7FFFu + ((v.u >> 16) & 1u);   // round-to-nearest-even
  unsigned short h = (unsigned short)(r >> 16);
  return __builtin_bit_cast(bf16, h);
}
__device__ inline bf16 toBf(float v) { return f2bf(v); }
__device__ inline bf16 toBf(bf16 v)  { return v; }

__device__ inline v16bf mk16(v8bf lo, v8bf hi) {
  return __builtin_shufflevector(lo, hi, 0,1,2,3,4,5,6,7,8,9,10,11,12,13,14,15);
}

// ---------------------------------------------------------------------------
// CDNA5 async global->LDS copy (ASYNCcnt path, CDNA5 ISA)
// ---------------------------------------------------------------------------
typedef __attribute__((address_space(3))) const void* lds_cptr;

__device__ inline unsigned ldsOff(const void* p) {
  return (unsigned)(unsigned long long)(lds_cptr)p;   // addrspacecast -> LDS byte offset
}
__device__ inline void async_copy_b128(unsigned ldsAddr, unsigned long long gbase,
                                       unsigned byteOff) {
  // GVS mode: LDS dest from the vdst VGPR; global address is SGPR base plus VGPR byte offset
  asm volatile("global_load_async_to_lds_b128 %0, %1, %2 offset:0"
               :
               : "v"(ldsAddr), "v"(byteOff), "s"(gbase)
               : "memory");
}
__device__ inline void wait_async0() {
  asm volatile("s_wait_asynccnt 0" ::: "memory");
}

// ---------------------------------------------------------------------------
// Weight conversion kernels
// ---------------------------------------------------------------------------
__global__ void cvt_bf16_kernel(const float* __restrict__ in, bf16* __restrict__ out, int count) {
  int i = blockIdx.x * 256 + threadIdx.x;
  if (i < count) out[i] = f2bf(in[i]);
}

// w_cat (9,128,3,3) -> A[16][1152] with k = tap*128 + ci, rows 9..15 zero
__global__ void cvt_wcat_kernel(const float* __restrict__ w, bf16* __restrict__ out) {
  int i = blockIdx.x * 256 + threadIdx.x;     // 16*1152
  if (i >= 16 * 1152) return;
  int t9 = i / 1152;
  int k  = i % 1152;
  int tap = k >> 7, ci = k & 127;
  float v = 0.f;
  if (t9 < 9) v = w[(size_t)(t9 * 128 + ci) * 9 + tap];
  out[i] = f2bf(v);
}

// ---------------------------------------------------------------------------
// Main GEMM: C[n][m][p] = sum_k A[m][k] * B[n][k][p]
// block tile 128x128, 8 waves (4x2), K step 32, bf16 WMMA w/ f32 accum
// A staged with global_load_async_to_lds_b128; B transposed with v8bf packing
// ---------------------------------------------------------------------------
template <typename TB>
__global__ __launch_bounds__(256)
void gemm_kernel(const bf16* __restrict__ A, const TB* __restrict__ B,
                 float* __restrict__ C, int M, int K, int N,
                 long bStrideN, long cStrideN) {
  const int n  = blockIdx.z;
  const int mB = blockIdx.y * 128;
  const int pB = blockIdx.x * 128;
  B += (size_t)n * bStrideN;
  C += (size_t)n * cStrideN;

  __shared__ bf16 As[128][40];   // pitch 40 bf16 (80B) -> conflict-free frags
  __shared__ bf16 Bs[128][40];   // stored transposed: Bs[p][k]

  const int t = threadIdx.x;
  const int wave = t >> 5, lane = t & 31;
  const int wm = wave >> 1, wn = wave & 1;      // 4 x 2 wave grid
  const int lhalf = lane >> 4, l16 = lane & 15;

  v8f acc[2][4] = {};

  for (int k0 = 0; k0 < K; k0 += 32) {
    // stage A: 128 rows x 32 k via async b128 copies (ASYNCcnt)
    {
      unsigned long long gbase =
          (unsigned long long)(const void*)(A + (size_t)mB * K + k0);
#pragma unroll
      for (int i = 0; i < 2; ++i) {
        int idx = t + i * 256;                  // 0..511
        int m = idx >> 2, koff = (idx & 3) * 8;
        async_copy_b128(ldsOff(&As[m][koff]), gbase,
                        (unsigned)((m * K + koff) * 2));
      }
    }
    // stage B transposed: thread owns (j, khalf) -> two ds_store_b128
    {
      int j = t & 127;
      int khalf = t >> 7;                        // 0 or 1 (16 k each)
      const TB* src = B + (size_t)(k0 + khalf * 16) * N + pB + j;
      v8bf lo, hi;
#pragma unroll
      for (int e = 0; e < 8; ++e) lo[e] = toBf(src[(size_t)e * N]);
#pragma unroll
      for (int e = 0; e < 8; ++e) hi[e] = toBf(src[(size_t)(e + 8) * N]);
      *(v8bf*)&Bs[j][khalf * 16]     = lo;
      *(v8bf*)&Bs[j][khalf * 16 + 8] = hi;
    }
    wait_async0();
    __syncthreads();

    v16bf aF[2], bF[4];
#pragma unroll
    for (int mt = 0; mt < 2; ++mt) {
      int row = wm * 32 + mt * 16 + l16;
      v8bf lo = *(const v8bf*)&As[row][8 * lhalf];
      v8bf hi = *(const v8bf*)&As[row][16 + 8 * lhalf];
      aF[mt] = mk16(lo, hi);
    }
#pragma unroll
    for (int nt = 0; nt < 4; ++nt) {
      int col = wn * 64 + nt * 16 + l16;
      v8bf lo = *(const v8bf*)&Bs[col][8 * lhalf];
      v8bf hi = *(const v8bf*)&Bs[col][16 + 8 * lhalf];
      bF[nt] = mk16(lo, hi);
    }
#pragma unroll
    for (int mt = 0; mt < 2; ++mt)
#pragma unroll
      for (int nt = 0; nt < 4; ++nt)
        acc[mt][nt] = __builtin_amdgcn_wmma_f32_16x16x32_bf16(
            false, aF[mt], false, bF[nt], (short)0, acc[mt][nt], false, false);
    __syncthreads();
  }

  // epilogue: D layout lanes 0-15 => M=r, lanes 16-31 => M=r+8, N=lane%16
#pragma unroll
  for (int mt = 0; mt < 2; ++mt)
#pragma unroll
    for (int nt = 0; nt < 4; ++nt)
#pragma unroll
      for (int r = 0; r < 8; ++r) {
        int m = mB + wm * 32 + mt * 16 + r + 8 * lhalf;
        int p = pB + wn * 64 + nt * 16 + l16;
        C[(size_t)m * N + p] = acc[mt][nt][r];
      }
}

// ---------------------------------------------------------------------------
// Branch conv GEMM: logits[b][n][16][p] = Wcat(16x1152) x im2col(xd, dil)
// K = tap*128 + ci ; per-32 K-chunk => single tap, contiguous ci slice.
// Whole A (36KB) staged once in LDS (async); im2col loads are clamped
// (unconditional, coalesced) then zero-selected -> no exec divergence.
// ---------------------------------------------------------------------------
__global__ __launch_bounds__(256)
void branch_gemm_kernel(const bf16* __restrict__ wcat,
                        const bf16* __restrict__ xd16,
                        float* __restrict__ logits) {
  const int pB = blockIdx.x * 256;
  const int b  = blockIdx.y;
  const int n  = blockIdx.z;
  const int dil = c_dils[b];

  __shared__ bf16 As[16][1160];   // full 16x1152 A, padded pitch
  __shared__ bf16 Bs[256][40];

  const int t = threadIdx.x;
  const int wave = t >> 5, lane = t & 31;
  const int lhalf = lane >> 4, l16 = lane & 15;
  const bf16 z = __builtin_bit_cast(bf16, (unsigned short)0);

  // ---- stage full A once: 16*1152 bf16 = 2304 x b128 async copies ----
  {
    unsigned long long gbase =
        (unsigned long long)(const void*)(wcat + (size_t)b * 16 * 1152);
#pragma unroll
    for (int i = 0; i < 9; ++i) {
      int idx = t + i * 256;                 // 0..2303
      int m = idx / 144, koff = (idx % 144) * 8;
      async_copy_b128(ldsOff(&As[m][koff]), gbase,
                      (unsigned)((m * 1152 + koff) * 2));
    }
  }
  wait_async0();
  __syncthreads();

  v8f acc[2] = {};

  const int j = t;                            // this thread's pixel column
  const int p = pB + j;
  const int py = p >> 7, px = p & 127;

  for (int kc = 0; kc < 36; ++kc) {           // 1152 / 32
    int k0  = kc * 32;
    int tap = kc >> 2;
    int ciB = (kc & 3) * 32;
    int py2 = py + c_ky[tap] * dil;
    int px2 = px + c_kx[tap] * dil;
    bool ok = ((unsigned)py2 < 128u) && ((unsigned)px2 < 128u);
    int pyc = py2 < 0 ? 0 : (py2 > 127 ? 127 : py2);
    int pxc = px2 < 0 ? 0 : (px2 > 127 ? 127 : px2);
    const bf16* src = xd16 + ((size_t)(n * CI + ciB)) * NPIX + pyc * 128 + pxc;

    // stage B: 32 shifted rows, unconditional clamped loads + select
#pragma unroll
    for (int g8 = 0; g8 < 4; ++g8) {
      v8bf tmp;
#pragma unroll
      for (int e = 0; e < 8; ++e) {
        bf16 vv = src[(size_t)(g8 * 8 + e) * NPIX];
        tmp[e] = ok ? vv : z;
      }
      *(v8bf*)&Bs[j][g8 * 8] = tmp;
    }
    __syncthreads();

    v16bf aF = mk16(*(const v8bf*)&As[l16][k0 + 8 * lhalf],
                    *(const v8bf*)&As[l16][k0 + 16 + 8 * lhalf]);
    v16bf bF[2];
#pragma unroll
    for (int nt = 0; nt < 2; ++nt) {
      int col = wave * 32 + nt * 16 + l16;
      bF[nt] = mk16(*(const v8bf*)&Bs[col][8 * lhalf],
                    *(const v8bf*)&Bs[col][16 + 8 * lhalf]);
    }
#pragma unroll
    for (int nt = 0; nt < 2; ++nt)
      acc[nt] = __builtin_amdgcn_wmma_f32_16x16x32_bf16(
          false, aF, false, bF[nt], (short)0, acc[nt], false, false);
    __syncthreads();
  }

  size_t base = ((size_t)(b * NB + n) * 16) * NPIX;
#pragma unroll
  for (int nt = 0; nt < 2; ++nt)
#pragma unroll
    for (int r = 0; r < 8; ++r) {
      int m = r + 8 * lhalf;
      int pp = pB + wave * 32 + nt * 16 + l16;
      logits[base + (size_t)m * NPIX + pp] = acc[nt][r];
    }
}

// ---------------------------------------------------------------------------
// Softmax over the 9 taps (in place on logits)
// ---------------------------------------------------------------------------
__global__ __launch_bounds__(256)
void softmax9_kernel(float* __restrict__ logits) {
  int i = blockIdx.x * 256 + threadIdx.x;      // 0 .. 4*4*16384-1
  int p  = i & (NPIX - 1);
  int bn = i >> 14;
  float* base = logits + ((size_t)bn * 16) * NPIX + p;
  float v[9], m = -1e30f;
#pragma unroll
  for (int t9 = 0; t9 < 9; ++t9) { v[t9] = base[(size_t)t9 * NPIX]; m = fmaxf(m, v[t9]); }
  float s = 0.f;
#pragma unroll
  for (int t9 = 0; t9 < 9; ++t9) { v[t9] = __expf(v[t9] - m); s += v[t9]; }
  float inv = 1.f / s;
#pragma unroll
  for (int t9 = 0; t9 < 9; ++t9) base[(size_t)t9 * NPIX] = v[t9] * inv;
}

// ---------------------------------------------------------------------------
// GroupNorm statistics: one block per (n, group) -> mean, inv_std (float4 loads)
// ---------------------------------------------------------------------------
__global__ __launch_bounds__(256)
void gn_stats_kernel(const float* __restrict__ raw, float* __restrict__ stats,
                     int cpg, long nStride) {
  int bid = blockIdx.x;
  int n = bid >> 5, g = bid & 31;
  const float* base = raw + (size_t)n * nStride + (size_t)g * cpg * NPIX;
  long cnt = (long)cpg * NPIX;
  long cnt4 = cnt >> 2;
  const v4f* b4 = (const v4f*)base;
  int t = threadIdx.x;
  float s = 0.f, s2 = 0.f;
  for (long i = t; i < cnt4; i += 256) {
    v4f v = b4[i];
#pragma unroll
    for (int e = 0; e < 4; ++e) { s += v[e]; s2 += v[e] * v[e]; }
  }
  __shared__ float ss[256], ss2[256];
  ss[t] = s; ss2[t] = s2; __syncthreads();
  for (int o = 128; o > 0; o >>= 1) {
    if (t < o) { ss[t] += ss[t + o]; ss2[t] += ss2[t + o]; }
    __syncthreads();
  }
  if (t == 0) {
    float mean = ss[0] / (float)cnt;
    float var  = ss2[0] / (float)cnt - mean * mean;
    stats[2 * bid]     = mean;
    stats[2 * bid + 1] = rsqrtf(var + 1e-5f);
  }
}

// ---------------------------------------------------------------------------
// GroupNorm apply + ReLU (4 px / thread), optional fp32 / bf16 / concat outputs
// ---------------------------------------------------------------------------
__global__ __launch_bounds__(256)
void gn_apply_kernel(const float* __restrict__ raw, const float* __restrict__ stats,
                     const float* __restrict__ gamma, const float* __restrict__ beta,
                     float* __restrict__ dstF, bf16* __restrict__ dstB,
                     bf16* __restrict__ dstCat, long catNStride, int catCOff,
                     int C, int cpg) {
  size_t i4 = ((size_t)blockIdx.x * 256 + threadIdx.x) * 4;   // over NB*C*NPIX
  int p = (int)(i4 & (NPIX - 1));
  long cn = (long)(i4 >> 14);
  int c = (int)(cn % C);
  int n = (int)(cn / C);
  int g = c / cpg;
  float mean = stats[2 * (n * 32 + g)];
  float is   = stats[2 * (n * 32 + g) + 1];
  float ga = gamma[c], be = beta[c];
  v4f r = *(const v4f*)&raw[i4];
  v4f v;
  v4bf vb;
#pragma unroll
  for (int e = 0; e < 4; ++e) {
    v[e]  = fmaxf((r[e] - mean) * is * ga + be, 0.f);
    vb[e] = f2bf(v[e]);
  }
  if (dstF) *(v4f*)&dstF[i4] = v;
  if (dstB) *(v4bf*)&dstB[i4] = vb;
  if (dstCat)
    *(v4bf*)&dstCat[(size_t)n * catNStride + (size_t)(catCOff + c) * NPIX + p] = vb;
}

// ---------------------------------------------------------------------------
// Weighted unfold: out[c][p] = sum_tap xd[c][p+shift(tap)] * f[tap][p]
// one block per (py, branch, n); filter row staged in LDS (L2-resident xd)
// ---------------------------------------------------------------------------
__global__ __launch_bounds__(128)
void unfold_kernel(const float* __restrict__ xd, const float* __restrict__ f,
                   bf16* __restrict__ cat) {
  int py = blockIdx.x, b = blockIdx.y, n = blockIdx.z;
  int px = threadIdx.x;
  int dil = c_dils[b];

  __shared__ float fs[9][128];
  size_t fbase = ((size_t)(b * NB + n) * 16) * NPIX + py * 128 + px;
#pragma unroll
  for (int t9 = 0; t9 < 9; ++t9) fs[t9][px] = f[fbase + (size_t)t9 * NPIX];
  __syncthreads();

  int  off[9];
  bool ok[9];
#pragma unroll
  for (int t9 = 0; t9 < 9; ++t9) {
    int py2 = py + c_ky[t9] * dil, px2 = px + c_kx[t9] * dil;
    ok[t9]  = ((unsigned)py2 < 128u) && ((unsigned)px2 < 128u);
    int pyc = py2 < 0 ? 0 : (py2 > 127 ? 127 : py2);
    int pxc = px2 < 0 ? 0 : (px2 > 127 ? 127 : px2);
    off[t9] = pyc * 128 + pxc;
  }
  const float* xn = xd + (size_t)n * CI * NPIX;
  bf16* outb = cat + (size_t)n * (5 * CI) * NPIX +
               (size_t)(1 + b) * CI * NPIX + py * 128 + px;
  for (int c = 0; c < CI; ++c) {
    const float* xc = xn + (size_t)c * NPIX;
    float s = 0.f;
#pragma unroll
    for (int t9 = 0; t9 < 9; ++t9) {
      float xv = xc[off[t9]];
      s += (ok[t9] ? xv : 0.f) * fs[t9][px];
    }
    outb[(size_t)c * NPIX] = f2bf(s);
  }
}

// ---------------------------------------------------------------------------
// Launcher
// ---------------------------------------------------------------------------
extern "C" void kernel_launch(void* const* d_in, const int* in_sizes, int n_in,
                              void* d_out, int out_size, void* d_ws, size_t ws_size,
                              hipStream_t stream) {
  const float* x    = (const float*)d_in[0];
  const float* w1   = (const float*)d_in[1];
  const float* g1g  = (const float*)d_in[2];
  const float* g1b  = (const float*)d_in[3];
  const float* wca[4] = {(const float*)d_in[4], (const float*)d_in[5],
                         (const float*)d_in[6], (const float*)d_in[7]};
  const float* w2   = (const float*)d_in[8];
  const float* g2g  = (const float*)d_in[9];
  const float* g2b  = (const float*)d_in[10];
  const float* w3   = (const float*)d_in[11];
  const float* g3g  = (const float*)d_in[12];
  const float* g3b  = (const float*)d_in[13];
  float* out = (float*)d_out;

  char* ws = (char*)d_ws;
  size_t off = 0;
  auto alloc = [&](size_t bytes) {
    size_t o = off; off = (off + bytes + 255) & ~(size_t)255; return o;
  };
  float* stats1  = (float*)(ws + alloc(256 * 4));
  float* stats2  = (float*)(ws + alloc(256 * 4));
  float* stats3  = (float*)(ws + alloc(256 * 4));
  bf16*  w1b     = (bf16*)(ws + alloc((size_t)CI * CH * 2));
  bf16*  w2b     = (bf16*)(ws + alloc((size_t)CI * 5 * CI * 2));
  bf16*  w3b     = (bf16*)(ws + alloc((size_t)CH * CI * 2));
  bf16*  wcatb   = (bf16*)(ws + alloc((size_t)4 * 16 * 1152 * 2));
  float* c1raw   = (float*)(ws + alloc((size_t)NB * CI * NPIX * 4));
  float* xdF     = (float*)(ws + alloc((size_t)NB * CI * NPIX * 4));
  bf16*  xdB     = (bf16*)(ws + alloc((size_t)NB * CI * NPIX * 2));
  float* logits  = (float*)(ws + alloc((size_t)4 * NB * 16 * NPIX * 4));
  bf16*  cat     = (bf16*)(ws + alloc((size_t)NB * 5 * CI * NPIX * 2));
  float* c2raw   = (float*)(ws + alloc((size_t)NB * CI * NPIX * 4));
  bf16*  y2b     = (bf16*)(ws + alloc((size_t)NB * CI * NPIX * 2));
  (void)ws_size; (void)n_in; (void)in_sizes; (void)out_size;

  // --- weight conversion ---
  cvt_bf16_kernel<<<(CI * CH + 255) / 256, 256, 0, stream>>>(w1, w1b, CI * CH);
  cvt_bf16_kernel<<<(CI * 5 * CI + 255) / 256, 256, 0, stream>>>(w2, w2b, CI * 5 * CI);
  cvt_bf16_kernel<<<(CH * CI + 255) / 256, 256, 0, stream>>>(w3, w3b, CH * CI);
  for (int b = 0; b < 4; ++b)
    cvt_wcat_kernel<<<(16 * 1152 + 255) / 256, 256, 0, stream>>>(
        wca[b], wcatb + (size_t)b * 16 * 1152);

  // --- stage 1: 1x1 conv 512->128 (WMMA GEMM) + GN + ReLU ---
  gemm_kernel<float><<<dim3(NPIX / 128, 1, NB), 256, 0, stream>>>(
      w1b, x, c1raw, CI, CH, NPIX, (long)CH * NPIX, (long)CI * NPIX);
  gn_stats_kernel<<<NB * GROUPS, 256, 0, stream>>>(c1raw, stats1, CI / GROUPS,
                                                   (long)CI * NPIX);
  gn_apply_kernel<<<(NB * CI * NPIX) / 1024, 256, 0, stream>>>(
      c1raw, stats1, g1g, g1b, xdF, xdB, cat, (long)5 * CI * NPIX, 0, CI,
      CI / GROUPS);

  // --- dynamic filter branches: dilated 3x3 conv (WMMA) + softmax + unfold ---
  branch_gemm_kernel<<<dim3(NPIX / 256, 4, NB), 256, 0, stream>>>(wcatb, xdB, logits);
  softmax9_kernel<<<(4 * NB * NPIX) / 256, 256, 0, stream>>>(logits);
  unfold_kernel<<<dim3(HW, 4, NB), 128, 0, stream>>>(xdF, logits, cat);

  // --- stage 2: 1x1 conv 640->128 + GN + ReLU ---
  gemm_kernel<bf16><<<dim3(NPIX / 128, 1, NB), 256, 0, stream>>>(
      w2b, cat, c2raw, CI, 5 * CI, NPIX, (long)5 * CI * NPIX, (long)CI * NPIX);
  gn_stats_kernel<<<NB * GROUPS, 256, 0, stream>>>(c2raw, stats2, CI / GROUPS,
                                                   (long)CI * NPIX);
  gn_apply_kernel<<<(NB * CI * NPIX) / 1024, 256, 0, stream>>>(
      c2raw, stats2, g2g, g2b, (float*)nullptr, y2b, (bf16*)nullptr, 0, 0, CI,
      CI / GROUPS);

  // --- stage 3: 1x1 conv 128->512 + GN + ReLU (raw to d_out, then in-place) ---
  gemm_kernel<bf16><<<dim3(NPIX / 128, CH / 128, NB), 256, 0, stream>>>(
      w3b, y2b, out, CH, CI, NPIX, (long)CI * NPIX, (long)CH * NPIX);
  gn_stats_kernel<<<NB * GROUPS, 256, 0, stream>>>(out, stats3, CH / GROUPS,
                                                   (long)CH * NPIX);
  gn_apply_kernel<<<((size_t)NB * CH * NPIX) / 1024, 256, 0, stream>>>(
      out, stats3, g3g, g3b, out, (bf16*)nullptr, (bf16*)nullptr, 0, 0, CH,
      CH / GROUPS);
}
